// MambaLayer_47897475285516
// MI455X (gfx1250) — compile-verified, matrix operand order checked by reading, not run on hardware
//
#include <hip/hip_runtime.h>
#include <math.h>

// ---------------- problem constants (from reference) ----------------
constexpr int BB     = 4;
constexpr int DIM    = 192;
constexpr int L      = 2048;           // 16*16*8
constexpr int DINNER = 384;            // EXPAND * DIM
constexpr int NXZ    = 2 * DINNER;     // 768
constexpr int DTR    = 12;             // dt_rank
constexpr int DST    = 16;             // d_state
constexpr int XPO    = DTR + 2 * DST;  // 44
constexpr int TOK    = BB * L;         // 8192 tokens

// ---------------- bf16 helpers ----------------
static __device__ __forceinline__ unsigned short f2bf(float f) {
    unsigned int u = __builtin_bit_cast(unsigned int, f);
    u += 0x7FFFu + ((u >> 16) & 1u);   // round-to-nearest-even
    return (unsigned short)(u >> 16);
}

typedef __attribute__((ext_vector_type(16))) __bf16 v16bf;
typedef __attribute__((ext_vector_type(8)))  float  v8f;

union FragAB { unsigned int u[8]; v16bf bf; };
union FragC  { v8f v; float f[8]; };

// ---------------- weight f32 -> bf16 ----------------
__global__ void k_f32_to_bf16(const float* __restrict__ src,
                              unsigned short* __restrict__ dst, int n) {
    int i = blockIdx.x * blockDim.x + threadIdx.x;
    if (i < n) dst[i] = f2bf(src[i]);
}

// ---------------- LayerNorm over DIM, emit bf16 tokens ----------------
// x layout: (B, DIM, L) ; output xn: (TOK, DIM) row-major bf16
__global__ void k_layernorm(const float* __restrict__ x,
                            const float* __restrict__ w,
                            const float* __restrict__ bn,
                            unsigned short* __restrict__ xn) {
    int t = blockIdx.x * blockDim.x + threadIdx.x;
    if (t >= TOK) return;
    int b = t >> 11, l = t & (L - 1);
    const float* xb = x + (size_t)b * DIM * L + l;   // stride L per channel
    float s = 0.f, s2 = 0.f;
    for (int c = 0; c < DIM; ++c) {
        float v = xb[(size_t)c * L];
        s += v; s2 += v * v;
    }
    float mu  = s * (1.f / DIM);
    float var = s2 * (1.f / DIM) - mu * mu;
    float inv = rsqrtf(var + 1e-5f);
    unsigned short* o = xn + (size_t)t * DIM;
    for (int c = 0; c < DIM; ++c) {
        float v = (xb[(size_t)c * L] - mu) * inv * w[c] + bn[c];
        o[c] = f2bf(v);
    }
}

// ---------------- fragment load: A/B operand, lane&15 = row/col ------------
// 8 VGPRs of bf16 pairs; k = ((v&4)<<2) + 8*lane_hi + 2*(v&3)  (ISA 7.12.2)
static __device__ __forceinline__ FragAB load_frag(const unsigned short* __restrict__ row,
                                                   int kk, int lh) {
    FragAB f;
#pragma unroll
    for (int v = 0; v < 8; ++v) {
        int kp = ((v & 4) << 2) + lh * 8 + (v & 3) * 2;
        f.u[v] = *(const unsigned int*)(row + kk + kp);
    }
    return f;
}

// ---------------- GEMM1: xz[TOK,768] = xn[TOK,192] @ W_in[768,192]^T -------
// One wave computes a 16x64 strip (4 accumulators): A fragment loaded once
// per k-step and reused across 4 v_wmma — 4x lower activation traffic.
__global__ void k_gemm_xz(const unsigned short* __restrict__ A,
                          const unsigned short* __restrict__ Bw,
                          float* __restrict__ C) {
    constexpr int K = DIM, N = NXZ;
    constexpr int ntN = N / 16;          // 48
    constexpr int strips = (TOK / 16) * (ntN / 4);   // 512*12 = 6144 waves
    int wave = (int)((blockIdx.x * blockDim.x + threadIdx.x) >> 5);
    int lane = threadIdx.x & 31;
    if (wave >= strips) return;
    int tM  = wave / (ntN / 4);
    int tN0 = (wave % (ntN / 4)) * 4;
    int lh  = (lane >> 4) & 1;
    const unsigned short* Arow  = A  + (size_t)(tM * 16 + (lane & 15)) * K;
    const unsigned short* Brow0 = Bw + (size_t)(tN0 * 16 + (lane & 15)) * K;
    v8f acc[4] = {};
#pragma unroll
    for (int kk = 0; kk < K; kk += 32) {
        if (kk + 32 < K) __builtin_prefetch(Arow + kk + 32, 0, 0);
        FragAB a = load_frag(Arow, kk, lh);
#pragma unroll
        for (int j = 0; j < 4; ++j) {
            FragAB b = load_frag(Brow0 + (size_t)j * 16 * K, kk, lh);
            acc[j] = __builtin_amdgcn_wmma_f32_16x16x32_bf16(
                         false, a.bf, false, b.bf, (short)0, acc[j], false, false);
        }
    }
#pragma unroll
    for (int j = 0; j < 4; ++j) {
        FragC c; c.v = acc[j];
#pragma unroll
        for (int v = 0; v < 8; ++v) {
            int row = tM * 16 + v + 8 * lh;          // token
            C[(size_t)row * N + (tN0 + j) * 16 + (lane & 15)] = c.f[v];
        }
    }
}

// ---------------- GEMM2: out = y[TOK,384] @ W_out[192,384]^T, store (B,C,L) -
__global__ void k_gemm_out(const unsigned short* __restrict__ A,
                           const unsigned short* __restrict__ Bw,
                           float* __restrict__ out) {
    constexpr int K = DINNER, N = DIM;
    constexpr int ntN = N / 16;          // 12
    constexpr int strips = (TOK / 16) * (ntN / 4);   // 512*3 = 1536 waves
    int wave = (int)((blockIdx.x * blockDim.x + threadIdx.x) >> 5);
    int lane = threadIdx.x & 31;
    if (wave >= strips) return;
    int tM  = wave / (ntN / 4);
    int tN0 = (wave % (ntN / 4)) * 4;
    int lh  = (lane >> 4) & 1;
    const unsigned short* Arow  = A  + (size_t)(tM * 16 + (lane & 15)) * K;
    const unsigned short* Brow0 = Bw + (size_t)(tN0 * 16 + (lane & 15)) * K;
    v8f acc[4] = {};
#pragma unroll
    for (int kk = 0; kk < K; kk += 32) {
        if (kk + 32 < K) __builtin_prefetch(Arow + kk + 32, 0, 0);
        FragAB a = load_frag(Arow, kk, lh);
#pragma unroll
        for (int j = 0; j < 4; ++j) {
            FragAB b = load_frag(Brow0 + (size_t)j * 16 * K, kk, lh);
            acc[j] = __builtin_amdgcn_wmma_f32_16x16x32_bf16(
                         false, a.bf, false, b.bf, (short)0, acc[j], false, false);
        }
    }
#pragma unroll
    for (int j = 0; j < 4; ++j) {
        FragC c; c.v = acc[j];
        int col = (tN0 + j) * 16 + (lane & 15);      // output channel
#pragma unroll
        for (int v = 0; v < 8; ++v) {
            int token = tM * 16 + v + 8 * lh;
            int b = token >> 11, l = token & (L - 1);
            out[((size_t)b * DIM + col) * L + l] = c.f[v];   // (B, C, img)
        }
    }
}

// ---------------- causal depthwise conv (k=4) + SiLU -----------------------
__global__ void k_conv(const float* __restrict__ xz,
                       const float* __restrict__ Wc,
                       const float* __restrict__ bc,
                       float* __restrict__ xc) {
    int i = blockIdx.x * blockDim.x + threadIdx.x;
    if (i >= TOK * DINNER) return;
    int d = i % DINNER, t = i / DINNER;
    int b = t >> 11, l = t & (L - 1);
    float acc = bc[d];
#pragma unroll
    for (int j = 0; j < 4; ++j) {
        int li = l - 3 + j;
        if (li >= 0)
            acc += Wc[d * 4 + j] * xz[(size_t)(b * L + li) * NXZ + d];
    }
    xc[(size_t)t * DINNER + d] = acc / (1.f + __expf(-acc));   // SiLU
}

// ---------------- x-proj (44) + delta-proj (384) + softplus ----------------
__global__ void k_xproj(const float* __restrict__ xc,
                        const float* __restrict__ Wxp,
                        const float* __restrict__ Wdt,
                        const float* __restrict__ bdt,
                        float* __restrict__ delta,
                        float* __restrict__ Bs,
                        float* __restrict__ Cs) {
    __shared__ float sx[DINNER];
    __shared__ float sd[XPO];
    int t = blockIdx.x, tid = threadIdx.x;
    for (int i = tid; i < DINNER; i += blockDim.x) sx[i] = xc[(size_t)t * DINNER + i];
    __syncthreads();
    if (tid < XPO) {
        const float* w = Wxp + tid * DINNER;
        float acc = 0.f;
        for (int k = 0; k < DINNER; ++k) acc += w[k] * sx[k];
        sd[tid] = acc;
    }
    __syncthreads();
    if (tid < DST) {
        Bs[(size_t)t * DST + tid] = sd[DTR + tid];
        Cs[(size_t)t * DST + tid] = sd[DTR + DST + tid];
    }
    for (int d = tid; d < DINNER; d += blockDim.x) {
        float acc = bdt[d];
#pragma unroll
        for (int r = 0; r < DTR; ++r) acc += Wdt[d * DTR + r] * sd[r];
        delta[(size_t)t * DINNER + d] = (acc > 20.f) ? acc : log1pf(__expf(acc));
    }
}

// ---------------- selective scan: 1536 independent (b,d) recurrences -------
// fuses D_skip add + SiLU(z) gate, emits bf16 rows for the out-proj GEMM
__global__ void k_scan(const float* __restrict__ delta,
                       const float* __restrict__ xc,
                       const float* __restrict__ Bs,
                       const float* __restrict__ Cs,
                       const float* __restrict__ xz,
                       const float* __restrict__ Alog,
                       const float* __restrict__ Dsk,
                       unsigned short* __restrict__ ybf) {
    int gid = blockIdx.x * blockDim.x + threadIdx.x;  // 0 .. BB*DINNER-1
    int b = gid / DINNER, d = gid % DINNER;
    float a[DST], h[DST];
#pragma unroll
    for (int n = 0; n < DST; ++n) {
        a[n] = -__expf(Alog[d * DST + n]);            // A = -exp(A_log)
        h[n] = 0.f;
    }
    float dski = Dsk[d];
    for (int l = 0; l < L; ++l) {
        size_t t = (size_t)b * L + l;
        float dl = delta[t * DINNER + d];
        float xv = xc[t * DINNER + d];
        float dx = dl * xv;
        const float* bp = Bs + t * DST;
        const float* cp = Cs + t * DST;
        float y = 0.f;
#pragma unroll
        for (int n = 0; n < DST; ++n) {
            h[n] = __expf(dl * a[n]) * h[n] + dx * bp[n];
            y += h[n] * cp[n];
        }
        y += dski * xv;
        float z = xz[t * NXZ + DINNER + d];           // gate branch
        float g = z / (1.f + __expf(-z));             // SiLU(z)
        ybf[t * DINNER + d] = f2bf(y * g);
    }
}

// ---------------- host launch ----------------
extern "C" void kernel_launch(void* const* d_in, const int* in_sizes, int n_in,
                              void* d_out, int out_size, void* d_ws, size_t ws_size,
                              hipStream_t stream) {
    (void)in_sizes; (void)n_in; (void)out_size; (void)ws_size;
    const float* x       = (const float*)d_in[0];
    const float* w_norm  = (const float*)d_in[1];
    const float* b_norm  = (const float*)d_in[2];
    const float* W_in    = (const float*)d_in[3];
    const float* W_conv  = (const float*)d_in[4];
    const float* b_conv  = (const float*)d_in[5];
    const float* W_xproj = (const float*)d_in[6];
    const float* W_dt    = (const float*)d_in[7];
    const float* b_dt    = (const float*)d_in[8];
    const float* A_log   = (const float*)d_in[9];
    const float* D_skip  = (const float*)d_in[10];
    const float* W_out   = (const float*)d_in[11];
    float* out = (float*)d_out;

    char* ws = (char*)d_ws;
    size_t off = 0;
    auto take = [&](size_t bytes) -> void* {
        void* p = ws + off;
        off = (off + bytes + 255) & ~(size_t)255;
        return p;
    };
    unsigned short* win_bf  = (unsigned short*)take((size_t)NXZ * DIM * 2);
    unsigned short* wout_bf = (unsigned short*)take((size_t)DIM * DINNER * 2);
    unsigned short* xn_bf   = (unsigned short*)take((size_t)TOK * DIM * 2);
    float*          xzf     = (float*)take((size_t)TOK * NXZ * 4);
    float*          xcf     = (float*)take((size_t)TOK * DINNER * 4);
    float*          deltaf  = (float*)take((size_t)TOK * DINNER * 4);
    float*          Bsf     = (float*)take((size_t)TOK * DST * 4);
    float*          Csf     = (float*)take((size_t)TOK * DST * 4);
    unsigned short* y_bf    = (unsigned short*)take((size_t)TOK * DINNER * 2);

    k_f32_to_bf16<<<(NXZ * DIM + 255) / 256, 256, 0, stream>>>(W_in, win_bf, NXZ * DIM);
    k_f32_to_bf16<<<(DIM * DINNER + 255) / 256, 256, 0, stream>>>(W_out, wout_bf, DIM * DINNER);
    k_layernorm<<<TOK / 256, 256, 0, stream>>>(x, w_norm, b_norm, xn_bf);
    // 6144 strip-waves / 8 waves per block = 768 blocks
    k_gemm_xz<<<(TOK / 16) * (NXZ / 64) / 8, 256, 0, stream>>>(xn_bf, win_bf, xzf);
    k_conv<<<(TOK * DINNER) / 256, 256, 0, stream>>>(xzf, W_conv, b_conv, xcf);
    k_xproj<<<TOK, 128, 0, stream>>>(xcf, W_xproj, W_dt, b_dt, deltaf, Bsf, Csf);
    k_scan<<<(BB * DINNER) / 128, 128, 0, stream>>>(deltaf, xcf, Bsf, Csf, xzf,
                                                    A_log, D_skip, y_bf);
    // 1536 strip-waves / 8 = 192 blocks
    k_gemm_out<<<(TOK / 16) * (DIM / 64) / 8, 256, 0, stream>>>(y_bf, wout_bf, out);
}
